// SepSTSBlock_63745904607289
// MI455X (gfx1250) — matrix operand order repaired and checked by reading, not compiled, not run on hardware
//
#include <hip/hip_runtime.h>
#include <hip/hip_bf16.h>

// ---------------------------------------------------------------------------
// SepSTS block (Swin-3D style) for MI455X / gfx1250, wave32 + WMMA bf16.
//   1. ln_kernel(roll=1):  h0 = LN1(x) gathered at rolled coords        (bf16)
//   2. gemm: qkv_s = h0 @ d_qkv_w^T + b                                  (bf16)
//   3. attn_spatial: windowed MHA w/ rel-pos bias + shift mask -> aout   (bf16)
//   4. gemm(remap): h1 = unroll( aout @ d_proj_w^T + b )                 (bf16)
//   5. gemm: qkv_t = h1 @ p_qkv_w^T + b                                  (bf16)
//   6. attn_temporal (N=4, VALU) -> aout                                 (bf16)
//   7. gemm(resid): d_out = x + aout @ p_proj_w^T + b                    (f32)
//   8. ln_kernel(roll=0): ln2 = LN2(d_out)                               (bf16)
//   9. gemm(gelu): hmid = GELU(ln2 @ mlp_w1^T + b1)                      (bf16)
//  10. gemm(acc):  d_out += hmid @ mlp_w2^T + b2                         (f32)
// GEMM tiles are double-buffered in LDS: tile k+1 is staged with CDNA5
// async-to-LDS loads (ASYNCcnt) / Tensor Data Mover (TENSORcnt) while the
// WMMAs consume tile k; waits use asynccnt<=1 / tensorcnt<=1 to keep one
// transfer in flight.  Fallbacks keep the file compiling on any toolchain.
// ---------------------------------------------------------------------------

typedef __bf16 bf16;
typedef __attribute__((ext_vector_type(16))) __bf16 v16bf;
typedef __attribute__((ext_vector_type(8)))  float  v8f;
typedef __attribute__((ext_vector_type(4)))  int    v4i;
typedef __attribute__((ext_vector_type(4)))  unsigned int u32x4;
typedef __attribute__((ext_vector_type(8)))  int i32x8;
typedef __attribute__((ext_vector_type(4)))  int i32x4;

static constexpr int kB = 2, kD = 4, kH = 128, kW = 128, kC = 256;
static constexpr int kM = kB * kD * kH * kW;   // 131072 tokens
static constexpr int kNH = 8, kHD = 32, kHID = 1024;

// --------------------- CDNA5 async / TDM staging helpers -------------------
#if __has_builtin(__builtin_amdgcn_global_load_async_to_lds_b128)
#define HAS_ASYNC_LDS 1
__device__ __forceinline__ void async_copy16(void* lds_dst, const void* gsrc) {
  __builtin_amdgcn_global_load_async_to_lds_b128(
      (__attribute__((address_space(1))) v4i*)gsrc,
      (__attribute__((address_space(3))) v4i*)lds_dst, 0, 0);
}
#endif

template <int N>
__device__ __forceinline__ void wait_async() {
#if __has_builtin(__builtin_amdgcn_s_wait_asynccnt)
  __builtin_amdgcn_s_wait_asynccnt(N);
#else
  if (N == 0)
    asm volatile("s_wait_asynccnt 0" ::: "memory");
  else
    asm volatile("s_wait_asynccnt 1" ::: "memory");
#endif
}

template <int N>
__device__ __forceinline__ void wait_tensor() {
#if __has_builtin(__builtin_amdgcn_s_wait_tensorcnt)
  __builtin_amdgcn_s_wait_tensorcnt(N);
#else
  if (N == 0)
    asm volatile("s_wait_tensorcnt 0" ::: "memory");
  else
    asm volatile("s_wait_tensorcnt 1" ::: "memory");
#endif
}

__device__ __forceinline__ unsigned lds_offset_of(const void* p) {
  return (unsigned)(unsigned long long)
      (__attribute__((address_space(3))) const char*)p;
}

#if __has_builtin(__builtin_amdgcn_tensor_load_to_lds)
#define HAS_TDM 1
// DMA a [rows x cols] tile of a row-major 16-bit tensor (row stride
// `stride_elems`) into LDS at byte offset `ldsoff`, inserting a 16B pad after
// every 64B of data (-> 40-half LDS row stride for cols==32).
__device__ __forceinline__ void tdm_load_tile_b16(const void* gbase,
                                                  unsigned ldsoff, int cols,
                                                  int rows, int stride_elems,
                                                  int tensor_rows) {
  const unsigned long long ga = (unsigned long long)gbase;
  u32x4 g0;
  g0[0] = 1u;                                   // count=1 valid descriptor
  g0[1] = ldsoff;                               // lds_addr (bytes)
  g0[2] = (unsigned)(ga & 0xffffffffu);         // global_addr[31:0]
  g0[3] = (unsigned)((ga >> 32) & 0x01ffffffu)  // global_addr[56:32]
          | 0x80000000u;                        // type = 2 ("image")
  i32x8 g1;
  g1[0] = (1 << 16)    // data_size = 2 bytes
        | (1 << 20)    // pad_enable
        | (3 << 22)    // pad_interval: 16 DWORDs (64B) of data per pad
        | (3 << 25);   // pad_amount:   4 DWORDs (16B) of pad
  g1[1] = (stride_elems & 0xffff) << 16;                       // dim0 lo16
  g1[2] = (int)(((unsigned)stride_elems >> 16) |
                (((unsigned)tensor_rows & 0xffffu) << 16));    // dim0 hi/dim1 lo
  g1[3] = (int)(((unsigned)tensor_rows >> 16) |
                ((unsigned)cols << 16));                       // tile_dim0
  g1[4] = rows;                                                // tile_dim1
  g1[5] = stride_elems;                                        // dim0 stride
  g1[6] = 0;
  g1[7] = 0;
  const i32x4 gz = {0, 0, 0, 0};
#if __clang_major__ >= 23
  const i32x8 gz8 = {0, 0, 0, 0, 0, 0, 0, 0};
  __builtin_amdgcn_tensor_load_to_lds(g0, g1, gz, gz, gz8, 0);
#else
  __builtin_amdgcn_tensor_load_to_lds(g0, g1, gz, gz, 0);
#endif
}
#endif

// ------------------------- fragment builders (LDS) -------------------------
// A-matrix 16x32 bf16 layout: lane holds m=lane%16; slot i -> k =
//   16*(i>=8) + 8*(lane>=16) + (i%8)
__device__ __forceinline__ v16bf lds_frag_a(const bf16* __restrict__ s) {
  const int lane = threadIdx.x & 31;
  const int m = lane & 15, kh = (lane >> 4) << 3;
  v16bf f;
#pragma unroll
  for (int i = 0; i < 16; ++i)
    f[i] = s[m * 40 + ((i >= 8) ? 16 : 0) + kh + (i & 7)];
  return f;
}

// B-matrix 32x16 bf16 from weight rows (Wt is [N,K] row-major, B(k,n)=Wt[n][k]):
// lane holds n=lane%16; slot i -> k = 16*(i>=8) + 8*(lane>=16) + (i%8)
__device__ __forceinline__ v16bf lds_frag_b(const bf16* __restrict__ s) {
  const int lane = threadIdx.x & 31;
  const int n = lane & 15, kh = (lane >> 4) << 3;
  v16bf f;
#pragma unroll
  for (int i = 0; i < 16; ++i)
    f[i] = s[n * 40 + ((i >= 8) ? 16 : 0) + kh + (i & 7)];
  return f;
}

// ------------------------------ GEMM (WMMA) --------------------------------
// Out[M,N] = epi( A[M,K] @ Wt[N,K]^T + bias )
// EPI: 0 = bf16 store, 1 = exact-GELU bf16 store, 2 = f32 store of Res+acc,
//      3 = f32 accumulate into Out.  REMAP: roll-back scatter on row index.
template <int EPI, bool REMAP>
__global__ __launch_bounds__(256, 2) void gemm_wmma(
    const bf16* __restrict__ A, const bf16* __restrict__ Wt,
    const float* __restrict__ bias, void* __restrict__ OutP,
    const float* __restrict__ Res, int M, int N, int K) {
  __shared__ bf16 As[2][64 * 40];    // double-buffered 64x32 tiles
  __shared__ bf16 Bs[2][128 * 40];   // double-buffered 128x32 tiles

  const int tid = threadIdx.x;
  const int lane = tid & 31, wid = tid >> 5;
  const int bm = blockIdx.y * 64, bn = blockIdx.x * 128;
  const int wm = (wid >> 2) * 32, wn = (wid & 3) * 32;

  v8f acc[2][2];
#pragma unroll
  for (int i = 0; i < 2; ++i)
#pragma unroll
    for (int j = 0; j < 2; ++j)
#pragma unroll
      for (int r = 0; r < 8; ++r) acc[i][j][r] = 0.f;

  // stage one K-tile (32 wide) into LDS buffer `buf`
  auto stage = [&](int buf, int k0) {
    {  // A tile: 64x32 halfs = 256 x 16B chunks, one per thread
      const int r = tid >> 2, c = (tid & 3) * 8;
      const bf16* gsrc = A + (size_t)(bm + r) * K + k0 + c;
#if defined(HAS_ASYNC_LDS)
      async_copy16(&As[buf][r * 40 + c], gsrc);     // global -> LDS, ASYNCcnt
#else
      *(float4*)(&As[buf][r * 40 + c]) = *(const float4*)gsrc;
#endif
    }
#if defined(HAS_TDM)
    if (wid == 0) {  // one TDM descriptor moves the whole padded B tile
      tdm_load_tile_b16(Wt + (size_t)bn * K + k0, lds_offset_of(&Bs[buf][0]),
                        32, 128, K, N);
    }
#else
#pragma unroll
    for (int l = 0; l < 2; ++l) {  // B tile: 128x32 halfs, two chunks/thread
      const int idx = tid + l * 256;
      const int r = idx >> 2, c = (idx & 3) * 8;
      const bf16* gsrc = Wt + (size_t)(bn + r) * K + k0 + c;
#if defined(HAS_ASYNC_LDS)
      async_copy16(&Bs[buf][r * 40 + c], gsrc);
#else
      *(float4*)(&Bs[buf][r * 40 + c]) = *(const float4*)gsrc;
#endif
    }
#endif
  };

  const int nsteps = K >> 5;
  stage(0, 0);  // prologue: tile 0 in flight

#pragma unroll 2
  for (int kt = 0; kt < nsteps; ++kt) {
    const int cur = kt & 1;
    const bool have_next = (kt + 1) < nsteps;
    if (have_next) stage(cur ^ 1, (kt + 1) << 5);  // issue next tile's DMA
    if (kt + 2 < nsteps) {  // global_prefetch_b8 of the tile after next
      __builtin_prefetch(A + (size_t)(bm + (tid >> 2)) * K + ((kt + 2) << 5),
                         0, 1);
      __builtin_prefetch(Wt + (size_t)(bn + (tid >> 1)) * K + ((kt + 2) << 5),
                         0, 1);
    }
    // wait for the CURRENT tile only; the next tile stays in flight
#if defined(HAS_TDM)
    if (wid == 0) {
      if (have_next) wait_tensor<1>(); else wait_tensor<0>();
    }
#endif
#if defined(HAS_ASYNC_LDS)
    if (have_next) wait_async<1>(); else wait_async<0>();
#endif
    __syncthreads();

    const v16bf a0 = lds_frag_a(&As[cur][wm * 40]);
    const v16bf a1 = lds_frag_a(&As[cur][(wm + 16) * 40]);
    const v16bf b0 = lds_frag_b(&Bs[cur][wn * 40]);
    const v16bf b1 = lds_frag_b(&Bs[cur][(wn + 16) * 40]);
    acc[0][0] = __builtin_amdgcn_wmma_f32_16x16x32_bf16(
        false, a0, false, b0, (short)0, acc[0][0], false, false);
    acc[0][1] = __builtin_amdgcn_wmma_f32_16x16x32_bf16(
        false, a0, false, b1, (short)0, acc[0][1], false, false);
    acc[1][0] = __builtin_amdgcn_wmma_f32_16x16x32_bf16(
        false, a1, false, b0, (short)0, acc[1][0], false, false);
    acc[1][1] = __builtin_amdgcn_wmma_f32_16x16x32_bf16(
        false, a1, false, b1, (short)0, acc[1][1], false, false);
    __syncthreads();  // buffer `cur` may be overwritten at kt+2
  }

  // epilogue: C/D layout m = r + 8*(lane>=16), n = lane%16
  const int nl = lane & 15, mh = (lane >> 4) * 8;
#pragma unroll
  for (int i = 0; i < 2; ++i) {
#pragma unroll
    for (int j = 0; j < 2; ++j) {
      const int col = bn + wn + j * 16 + nl;
      const float bv = bias ? bias[col] : 0.f;
#pragma unroll
      for (int r = 0; r < 8; ++r) {
        const int row = bm + wm + i * 16 + mh + r;
        float v = acc[i][j][r] + bv;
        size_t orow = (size_t)row;
        if (REMAP) {  // roll-back: rolled (h,w) lands at natural (h+4,w+4)
          const int w = row & 127, h = (row >> 7) & 127, img = row >> 14;
          orow = (size_t)img * 16384 + (size_t)((h + 4) & 127) * 128 +
                 ((w + 4) & 127);
        }
        const size_t idx = orow * (size_t)N + col;
        if (EPI == 0) {
          ((bf16*)OutP)[idx] = (bf16)v;
        } else if (EPI == 1) {
          const float g = 0.5f * v * (1.f + erff(v * 0.70710678118654752f));
          ((bf16*)OutP)[idx] = (bf16)g;
        } else if (EPI == 2) {
          ((float*)OutP)[idx] = Res[idx] + v;
        } else {
          ((float*)OutP)[idx] += v;
        }
      }
    }
  }
}

// ------------------------- spatial window attention ------------------------
// One block per 8x8 window (2048 windows), one wave per head.
// S = Q K^T via 16 WMMAs, softmax with bias+mask, P->LDS transpose, O = P V.
__global__ __launch_bounds__(256, 1) void attn_spatial(
    const bf16* __restrict__ qkv, const float* __restrict__ table,
    const float* __restrict__ mask, bf16* __restrict__ out) {
  __shared__ bf16 pbuf[8 * 64 * 64];  // 64 KB: per-head P (64x64)

  const int widx = blockIdx.x;
  const int head = threadIdx.x >> 5, lane = threadIdx.x & 31;
  const int wwin = widx & 15, hwin = (widx >> 4) & 15, img = widx >> 8;
  const int mwidx = widx & 1023;  // mask window index within one batch image
  bf16* pb = pbuf + head * 4096;

  const int nl = lane & 15, mh = (lane >> 4) * 8;
  const int kh8 = (lane >> 4) << 3, kh16 = (lane >> 4) << 4;

  auto rowof = [&](int t) -> size_t {
    return (size_t)img * 16384 + (size_t)(hwin * 8 + (t >> 3)) * 128 +
           (size_t)(wwin * 8 + (t & 7));
  };

  // ---- S = Q K^T  (M=64, N=64, K=32; 4x4 tiles) ----
  v8f s[4][4];
#pragma unroll
  for (int a = 0; a < 4; ++a)
#pragma unroll
    for (int b = 0; b < 4; ++b)
#pragma unroll
      for (int r = 0; r < 8; ++r) s[a][b][r] = 0.f;

  v16bf qa[4], kb[4];
#pragma unroll
  for (int mi = 0; mi < 4; ++mi) {
    const bf16* base = qkv + rowof(mi * 16 + nl) * 768 + head * 32;
#pragma unroll
    for (int i = 0; i < 16; ++i)
      qa[mi][i] = base[((i >= 8) ? 16 : 0) + kh8 + (i & 7)];
  }
#pragma unroll
  for (int nj = 0; nj < 4; ++nj) {
    const bf16* base = qkv + rowof(nj * 16 + nl) * 768 + 256 + head * 32;
#pragma unroll
    for (int i = 0; i < 16; ++i) kb[nj][i] = base[((i >= 8) ? 16 : 0) + kh8 + (i & 7)];
  }
#pragma unroll
  for (int mi = 0; mi < 4; ++mi)
#pragma unroll
    for (int nj = 0; nj < 4; ++nj)
      s[mi][nj] = __builtin_amdgcn_wmma_f32_16x16x32_bf16(
          false, qa[mi], false, kb[nj], (short)0, s[mi][nj], false, false);

  // ---- scale + rel-pos bias + shift mask + row softmax, P -> LDS ----
  const float scale = 0.17677669529663689f;  // 1/sqrt(32)
#pragma unroll
  for (int mi = 0; mi < 4; ++mi) {
#pragma unroll
    for (int r = 0; r < 8; ++r) {
      const int iq = mi * 16 + mh + r;
      const int hi = iq >> 3, wi = iq & 7;
      float v4[4], mx = -1e30f;
#pragma unroll
      for (int nj = 0; nj < 4; ++nj) {
        const int jq = nj * 16 + nl;
        const int rpi = (hi - (jq >> 3) + 7) * 15 + (wi - (jq & 7) + 7);
        const float val = s[mi][nj][r] * scale + table[rpi * 8 + head] +
                          mask[(size_t)mwidx * 4096 + iq * 64 + jq];
        v4[nj] = val;
        mx = fmaxf(mx, val);
      }
#pragma unroll
      for (int off = 1; off < 16; off <<= 1)
        mx = fmaxf(mx, __shfl_xor(mx, off, 16));
      float sum = 0.f;
#pragma unroll
      for (int nj = 0; nj < 4; ++nj) {
        v4[nj] = __expf(v4[nj] - mx);
        sum += v4[nj];
      }
#pragma unroll
      for (int off = 1; off < 16; off <<= 1) sum += __shfl_xor(sum, off, 16);
      const float inv = 1.f / sum;
#pragma unroll
      for (int nj = 0; nj < 4; ++nj)
        pb[iq * 64 + nj * 16 + nl] = (bf16)(v4[nj] * inv);
    }
  }

  // ---- O = P V  (M=64, N=32, K=64; two K-steps of 32) ----
  v8f o[4][2];
#pragma unroll
  for (int a = 0; a < 4; ++a)
#pragma unroll
    for (int b = 0; b < 2; ++b)
#pragma unroll
      for (int r = 0; r < 8; ++r) o[a][b][r] = 0.f;

#pragma unroll
  for (int kj = 0; kj < 2; ++kj) {
    v16bf pa[4];
#pragma unroll
    for (int mi = 0; mi < 4; ++mi)
#pragma unroll
      for (int i = 0; i < 16; ++i)
        pa[mi][i] = pb[(mi * 16 + nl) * 64 + kj * 32 + ((i >= 8) ? 16 : 0) +
                       kh8 + (i & 7)];
    v16bf vb[2];
#pragma unroll
    for (int n2 = 0; n2 < 2; ++n2)
#pragma unroll
      for (int i = 0; i < 16; ++i)
        vb[n2][i] = qkv[rowof(kj * 32 + ((i >= 8) ? 16 : 0) + kh8 + (i & 7)) * 768 + 512 + head * 32 +
                        n2 * 16 + nl];
#pragma unroll
    for (int mi = 0; mi < 4; ++mi)
#pragma unroll
      for (int n2 = 0; n2 < 2; ++n2)
        o[mi][n2] = __builtin_amdgcn_wmma_f32_16x16x32_bf16(
            false, pa[mi], false, vb[n2], (short)0, o[mi][n2], false, false);
  }

#pragma unroll
  for (int mi = 0; mi < 4; ++mi)
#pragma unroll
    for (int n2 = 0; n2 < 2; ++n2)
#pragma unroll
      for (int r = 0; r < 8; ++r)
        out[rowof(mi * 16 + mh + r) * 256 + head * 32 + n2 * 16 + nl] =
            (bf16)o[mi][n2][r];
}

// ----------------------- temporal attention (N = 4) ------------------------
// One thread per (group, head, query). 2*128*128 groups x 8 heads x 4 queries.
__global__ __launch_bounds__(256) void attn_temporal(
    const bf16* __restrict__ qkv, const float* __restrict__ table,
    bf16* __restrict__ out) {
  const int gid = blockIdx.x * 256 + threadIdx.x;
  if (gid >= kB * kH * kW * kNH * 4) return;
  const int qi = gid & 3;
  const int head = (gid >> 2) & 7;
  const int g = gid >> 5;
  const int w = g & 127, h = (g >> 7) & 127, b = g >> 14;

  auto rowof = [&](int dd) -> size_t {
    return (size_t)(b * 4 + dd) * 16384 + (size_t)h * 128 + (size_t)w;
  };

  const float scale = 0.17677669529663689f;
  float q[32];
  {
    const bf16* base = qkv + rowof(qi) * 768 + head * 32;
#pragma unroll
    for (int d = 0; d < 32; ++d) q[d] = (float)base[d];
  }
  float sc[4], mx = -1e30f;
#pragma unroll
  for (int j = 0; j < 4; ++j) {
    const bf16* base = qkv + rowof(j) * 768 + 256 + head * 32;
    float dot = 0.f;
#pragma unroll
    for (int d = 0; d < 32; ++d) dot += q[d] * (float)base[d];
    sc[j] = dot * scale + table[(qi - j + 3) * 8 + head];
    mx = fmaxf(mx, sc[j]);
  }
  float sum = 0.f;
#pragma unroll
  for (int j = 0; j < 4; ++j) {
    sc[j] = __expf(sc[j] - mx);
    sum += sc[j];
  }
  const float inv = 1.f / sum;
  float o[32];
#pragma unroll
  for (int d = 0; d < 32; ++d) o[d] = 0.f;
#pragma unroll
  for (int j = 0; j < 4; ++j) {
    const bf16* base = qkv + rowof(j) * 768 + 512 + head * 32;
    const float p = sc[j] * inv;
#pragma unroll
    for (int d = 0; d < 32; ++d) o[d] += p * (float)base[d];
  }
  bf16* ob = out + rowof(qi) * 256 + head * 32;
#pragma unroll
  for (int d = 0; d < 32; ++d) ob[d] = (bf16)o[d];
}

// ------------------------------- LayerNorm ---------------------------------
// One wave32 per token (C=256 -> 8 values/lane). roll!=0 gathers at (+4,+4).
__global__ __launch_bounds__(256) void ln_kernel(const float* __restrict__ x,
                                                 const float* __restrict__ g,
                                                 const float* __restrict__ b,
                                                 bf16* __restrict__ out,
                                                 int roll) {
  const int lane = threadIdx.x & 31, wid = threadIdx.x >> 5;
  const int tok = blockIdx.x * 8 + wid;
  size_t src = (size_t)tok;
  if (roll) {
    const int w = tok & 127, h = (tok >> 7) & 127, img = tok >> 14;
    src = (size_t)img * 16384 + (size_t)((h + 4) & 127) * 128 + ((w + 4) & 127);
  }
  const float* xr = x + src * 256;
  float v[8], sum = 0.f, sq = 0.f;
#pragma unroll
  for (int j = 0; j < 8; ++j) {
    v[j] = xr[lane + 32 * j];
    sum += v[j];
    sq += v[j] * v[j];
  }
#pragma unroll
  for (int off = 1; off < 32; off <<= 1) {
    sum += __shfl_xor(sum, off, 32);
    sq += __shfl_xor(sq, off, 32);
  }
  const float mu = sum * (1.f / 256.f);
  const float inv = rsqrtf(sq * (1.f / 256.f) - mu * mu + 1e-5f);
#pragma unroll
  for (int j = 0; j < 8; ++j) {
    const int c = lane + 32 * j;
    out[(size_t)tok * 256 + c] = (bf16)((v[j] - mu) * inv * g[c] + b[c]);
  }
}

// ----------------------------- f32 -> bf16 ---------------------------------
__global__ void cvt_bf16(const float* __restrict__ in, bf16* __restrict__ out,
                         int n) {
  const int i = blockIdx.x * 256 + threadIdx.x;
  if (i < n) out[i] = (bf16)in[i];
}

// ------------------------------- launcher ----------------------------------
extern "C" void kernel_launch(void* const* d_in, const int* in_sizes, int n_in,
                              void* d_out, int out_size, void* d_ws,
                              size_t ws_size, hipStream_t stream) {
  const float* x     = (const float*)d_in[0];
  const float* maskm = (const float*)d_in[1];
  const float* n1g   = (const float*)d_in[2];
  const float* n1b   = (const float*)d_in[3];
  const float* dqkvw = (const float*)d_in[4];
  const float* dqkvb = (const float*)d_in[5];
  const float* dprjw = (const float*)d_in[6];
  const float* dprjb = (const float*)d_in[7];
  const float* dtab  = (const float*)d_in[8];
  const float* pqkvw = (const float*)d_in[9];
  const float* pqkvb = (const float*)d_in[10];
  const float* pprjw = (const float*)d_in[11];
  const float* pprjb = (const float*)d_in[12];
  const float* ptab  = (const float*)d_in[13];
  const float* n2g   = (const float*)d_in[14];
  const float* n2b   = (const float*)d_in[15];
  const float* w1    = (const float*)d_in[16];
  const float* b1    = (const float*)d_in[17];
  const float* w2    = (const float*)d_in[18];
  const float* b2    = (const float*)d_in[19];
  float* out = (float*)d_out;

  // workspace carve-out (bf16 elements)
  bf16* wsb  = (bf16*)d_ws;
  bf16* wq_s = wsb;                 // 768*256
  bf16* wp_s = wq_s + 196608;       // 256*256
  bf16* wq_t = wp_s + 65536;        // 768*256
  bf16* wp_t = wq_t + 196608;       // 256*256
  bf16* ww1  = wp_t + 65536;        // 1024*256
  bf16* ww2  = ww1 + 262144;        // 256*1024
  bf16* h0   = ww2 + 262144;                  // M*256  (LN1 / LN2 out)
  bf16* big  = h0 + (size_t)kM * 256;         // M*1024 (qkv / mlp hidden)
  bf16* aout = big + (size_t)kM * 1024;       // M*256  (attention out)
  bf16* h1   = aout + (size_t)kM * 256;       // M*256  (post spatial proj)

  // weight conversions (f32 -> bf16, 2 MB total)
  cvt_bf16<<<768, 256, 0, stream>>>(dqkvw, wq_s, 196608);
  cvt_bf16<<<256, 256, 0, stream>>>(dprjw, wp_s, 65536);
  cvt_bf16<<<768, 256, 0, stream>>>(pqkvw, wq_t, 196608);
  cvt_bf16<<<256, 256, 0, stream>>>(pprjw, wp_t, 65536);
  cvt_bf16<<<1024, 256, 0, stream>>>(w1, ww1, 262144);
  cvt_bf16<<<1024, 256, 0, stream>>>(w2, ww2, 262144);

  // 1. LN1 with roll fold-in
  ln_kernel<<<kM / 8, 256, 0, stream>>>(x, n1g, n1b, h0, 1);
  // 2. spatial QKV
  gemm_wmma<0, false><<<dim3(6, kM / 64), 256, 0, stream>>>(
      h0, wq_s, dqkvb, big, nullptr, kM, 768, 256);
  // 3. spatial windowed attention
  attn_spatial<<<2048, 256, 0, stream>>>(big, dtab, maskm, aout);
  // 4. spatial proj + roll-back scatter
  gemm_wmma<0, true><<<dim3(2, kM / 64), 256, 0, stream>>>(
      aout, wp_s, dprjb, h1, nullptr, kM, 256, 256);
  // 5. temporal QKV
  gemm_wmma<0, false><<<dim3(6, kM / 64), 256, 0, stream>>>(
      h1, wq_t, pqkvb, big, nullptr, kM, 768, 256);
  // 6. temporal attention
  attn_temporal<<<4096, 256, 0, stream>>>(big, ptab, aout);
  // 7. temporal proj + residual -> f32 d_out
  gemm_wmma<2, false><<<dim3(2, kM / 64), 256, 0, stream>>>(
      aout, wp_t, pprjb, out, x, kM, 256, 256);
  // 8. LN2
  ln_kernel<<<kM / 8, 256, 0, stream>>>(out, n2g, n2b, h0, 0);
  // 9. MLP up + exact GELU
  gemm_wmma<1, false><<<dim3(8, kM / 64), 256, 0, stream>>>(
      h0, ww1, b1, big, nullptr, kM, 1024, 256);
  // 10. MLP down, accumulate into residual
  gemm_wmma<3, false><<<dim3(2, kM / 64), 256, 0, stream>>>(
      big, ww2, b2, out, nullptr, kM, 256, 1024);
}
